// MoeMLPMegaBlocksFixed_64321430225152
// MI455X (gfx1250) — matrix operand order repaired and checked by reading, not compiled, not run on hardware
//
#include <hip/hip_runtime.h>

// ============================================================================
// MoE MLP (top-2 of 8 experts) for MI455X / gfx1250.
// Routed (sparse) bf16 WMMA GEMMs: v_wmma_f32_16x16x32_bf16, f32 accumulate.
// Staging: async global->LDS copies (ASYNCcnt) w/ double-buffered LDS if the
// toolchain exposes the builtins, else register-pipelined staging.
// Workspace requirement: ~118 MB.
// ============================================================================

typedef __bf16 bf16;
typedef __attribute__((ext_vector_type(16))) __bf16 v16bf;
typedef __attribute__((ext_vector_type(8)))  __bf16 v8bf;
typedef __attribute__((ext_vector_type(4)))  __bf16 v4bf;
typedef __attribute__((ext_vector_type(8)))  float  v8f;
typedef __attribute__((ext_vector_type(4)))  int    v4i;

#define NTOK   8192
#define NEMBD  1024
#define NEXP   8
#define DFFN   2048
#define NW1COL (NEXP * DFFN)   // 16384

// GEMM tiling: 128x128 block tile, 8 waves of 32x64, K-step 32 (one WMMA K).
#define BM  128
#define BN  128
#define BK  32
#define LDA 40    // A LDS row stride (bf16): 80B rows -> conflict-free b128 frag reads
#define LDB 136   // B LDS row stride (bf16): 272B rows -> <=2-way conflicts

// Async staging path, guarded (probe-uncertain builtins).
#if defined(__has_builtin)
#  if __has_builtin(__builtin_amdgcn_global_load_async_to_lds_b128) && \
      __has_builtin(__builtin_amdgcn_s_wait_asynccnt)
#    define USE_ASYNC 1
#  endif
#endif
#ifndef USE_ASYNC
#  define USE_ASYNC 0
#endif

// ---- workspace layout (byte offsets) ----
static constexpr size_t OFF_XB  = 0;                                      // bf16 x        16 MB
static constexpr size_t OFF_W1B = OFF_XB  + (size_t)NTOK * NEMBD * 2;     // bf16 w1       32 MB
static constexpr size_t OFF_W2B = OFF_W1B + (size_t)NEMBD * NW1COL * 2;   // bf16 w2       32 MB
static constexpr size_t OFF_HB  = OFF_W2B + (size_t)NW1COL * NEMBD * 2;   // bf16 H        32 MB
static constexpr size_t OFF_CMB = OFF_HB  + (size_t)NTOK * DFFN * 2;      // f32 combine  256 KB
static constexpr size_t OFF_IDX = OFF_CMB + (size_t)NTOK * NEXP * 4;      // int idx      256 KB
static constexpr size_t OFF_CNT = OFF_IDX + (size_t)NEXP * NTOK * 4;      // int cnt[8]
static constexpr size_t OFF_FCN = OFF_CNT + 32;                           // int fcnt[8]
static constexpr size_t OFF_PRT = OFF_FCN + 32;                           // f32 partials 36 KB

// ---------------------------------------------------------------------------
__global__ void zero4_kernel(float4* __restrict__ p, int n4) {
  int i = blockIdx.x * 256 + threadIdx.x;
  if (i < n4) p[i] = make_float4(0.f, 0.f, 0.f, 0.f);
}

__global__ void cvt_bf16_kernel(const float4* __restrict__ src, v4bf* __restrict__ dst, int n4) {
  int i = blockIdx.x * 256 + threadIdx.x;
  if (i < n4) {
    float4 v = src[i];
    v4bf o;
    o[0] = (bf16)v.x; o[1] = (bf16)v.y; o[2] = (bf16)v.z; o[3] = (bf16)v.w;
    dst[i] = o;
  }
}

// ---------------------------------------------------------------------------
// Router: one wave32 per token. router_w cached in LDS (32 KB).
// ---------------------------------------------------------------------------
__global__ __launch_bounds__(256)
void router_kernel(const float* __restrict__ x, const float* __restrict__ rw,
                   float* __restrict__ comb, int* __restrict__ ecnt,
                   int* __restrict__ eidx, int* __restrict__ fcnt,
                   float* __restrict__ part)
{
  __shared__ float rws[NEXP * NEMBD];
  __shared__ float pblk[8][9];
  for (int i = threadIdx.x; i < NEXP * NEMBD; i += 256) rws[i] = rw[i];
  __syncthreads();

  const int wave = threadIdx.x >> 5, lane = threadIdx.x & 31;
  const int t = blockIdx.x * 8 + wave;

  float acc[NEXP];
#pragma unroll
  for (int e = 0; e < NEXP; ++e) acc[e] = 0.f;

  const float* xr = x + (size_t)t * NEMBD;
  for (int d = lane; d < NEMBD; d += 32) {
    float xv = xr[d];
#pragma unroll
    for (int e = 0; e < NEXP; ++e) acc[e] += xv * rws[e * NEMBD + d];
  }
#pragma unroll
  for (int off = 16; off >= 1; off >>= 1) {
#pragma unroll
    for (int e = 0; e < NEXP; ++e) acc[e] += __shfl_xor(acc[e], off, 32);
  }

  if (lane == 0) {
    float mx = acc[0];
#pragma unroll
    for (int e = 1; e < NEXP; ++e) mx = fmaxf(mx, acc[e]);
    float p[NEXP], s = 0.f;
#pragma unroll
    for (int e = 0; e < NEXP; ++e) { p[e] = expf(acc[e] - mx); s += p[e]; }
    float inv = 1.f / s;
#pragma unroll
    for (int e = 0; e < NEXP; ++e) p[e] *= inv;
    float z = mx + logf(s);                       // logsumexp(logits)

    int e1 = 0;
    for (int e = 1; e < NEXP; ++e) if (p[e] > p[e1]) e1 = e;   // ties -> lowest idx
    int e2 = (e1 == 0) ? 1 : 0;
    for (int e = 0; e < NEXP; ++e) if (e != e1 && p[e] > p[e2]) e2 = e;

    float den = p[e1] + p[e2];
    float cw1 = p[e1] / den, cw2 = p[e2] / den;
#pragma unroll
    for (int e = 0; e < NEXP; ++e)
      comb[(size_t)t * NEXP + e] = (e == e1) ? cw1 : ((e == e2) ? cw2 : 0.f);

    int p1 = atomicAdd(&ecnt[e1], 1); eidx[e1 * NTOK + p1] = t;
    int p2 = atomicAdd(&ecnt[e2], 1); eidx[e2 * NTOK + p2] = t;
    atomicAdd(&fcnt[e1], 1);
    atomicAdd(&fcnt[e2], 1);

#pragma unroll
    for (int e = 0; e < NEXP; ++e) pblk[wave][e] = p[e];
    pblk[wave][8] = z * z;
  }
  __syncthreads();
  if (threadIdx.x < 9) {
    float s = 0.f;
#pragma unroll
    for (int w = 0; w < 8; ++w) s += pblk[w][threadIdx.x];
    part[blockIdx.x * 9 + threadIdx.x] = s;       // deterministic block partial
  }
}

// Fixed-order reduction of partials -> z-loss, load-balance loss, f_i.
__global__ void finalize_kernel(const float* __restrict__ part, const int* __restrict__ fcnt,
                                float* __restrict__ tail)
{
  __shared__ float sums[9];
  if (threadIdx.x < 9) {
    float s = 0.f;
    for (int b = 0; b < NTOK / 8; ++b) s += part[b * 9 + threadIdx.x];
    sums[threadIdx.x] = s;
  }
  __syncthreads();
  if (threadIdx.x == 0) {
    float zloss = sums[8] * (1.f / NTOK);
    float lb = 0.f;
    float fi[NEXP];
#pragma unroll
    for (int e = 0; e < NEXP; ++e) {
      fi[e] = (float)fcnt[e] * (1.f / (NTOK * 2));
      lb += fi[e] * (sums[e] * (1.f / NTOK));
    }
    lb *= (float)NEXP;
    tail[0] = zloss;
    tail[1] = lb;
#pragma unroll
    for (int e = 0; e < NEXP; ++e) tail[2 + e] = fi[e];
  }
}

// ---------------------------------------------------------------------------
// Fragment loads.
// A frag (ISA 16-bit 16x32): lane L -> row L%16, K runs at (L/16)*8 and 16+(L/16)*8.
// B frag: lane L -> column N=L%16, K runs at (L/16)*8 and 16+(L/16)*8 (strided LDS rows).
// ---------------------------------------------------------------------------
__device__ __forceinline__ v16bf ld_frag_a(const bf16* p) {
  v8bf lo = *(const v8bf*)(p);
  v8bf hi = *(const v8bf*)(p + 16);
  v16bf r;
#pragma unroll
  for (int i = 0; i < 8; ++i) { r[i] = lo[i]; r[i + 8] = hi[i]; }
  return r;
}
__device__ __forceinline__ v16bf ld_frag_b(const bf16* p) {
  v16bf r;
#pragma unroll
  for (int i = 0; i < 8; ++i) { r[i] = p[i * LDB]; r[i + 8] = p[(16 + i) * LDB]; }
  return r;
}

// One K-step of WMMA for a wave's 32x64 sub-tile (2x4 tiles).
__device__ __forceinline__ void wmma_step(const bf16* __restrict__ Ab,
                                          const bf16* __restrict__ Bb,
                                          int wm, int wn, int lane,
                                          v8f acc[2][4])
{
  const int lrow = lane & 15, lhi = lane >> 4;
  v16bf af[2], bfm[4];
#pragma unroll
  for (int i = 0; i < 2; ++i)
    af[i] = ld_frag_a(Ab + (wm * 32 + i * 16 + lrow) * LDA + lhi * 8);
#pragma unroll
  for (int j = 0; j < 4; ++j)
    bfm[j] = ld_frag_b(Bb + lhi * 8 * LDB + wn * 64 + j * 16 + lrow);
#pragma unroll
  for (int i = 0; i < 2; ++i)
#pragma unroll
    for (int j = 0; j < 4; ++j)
      acc[i][j] = __builtin_amdgcn_wmma_f32_16x16x32_bf16(
          false, af[i], false, bfm[j], (short)0, acc[i][j], false, false);
}

#if USE_ASYNC
// Builtin signature (from hipcc diagnostic): arg0 = int4* addrspace(1) [global],
// arg1 = int4* addrspace(3) [LDS], arg2 = imm offset, arg3 = cpol.
typedef __attribute__((address_space(1))) v4i as1_v4i;
typedef __attribute__((address_space(3))) v4i as3_v4i;
__device__ __forceinline__ void async_cp16(const bf16* g, bf16* l) {
  __builtin_amdgcn_global_load_async_to_lds_b128(
      (as1_v4i*)g, (as3_v4i*)l, 0, 0);
}
#endif

// ---------------------------------------------------------------------------
// GEMM1 (per expert e): H = gelu(gather(x) @ W1_e), M = ecnt[e], N = 2048, K = 1024
// ---------------------------------------------------------------------------
__global__ __launch_bounds__(256)
void moe_gemm1_kernel(const bf16* __restrict__ xb, const bf16* __restrict__ w1b,
                      bf16* __restrict__ hb, const int* __restrict__ eidx,
                      const int* __restrict__ ecnt, int e)
{
  const int cnt = ecnt[e];
  const int m0  = blockIdx.y * BM;
  if (m0 >= cnt) return;                          // routed sparsity: skip empty row-blocks
  const int n0  = blockIdx.x * BN;

  __shared__ bf16 Alds[2][BM * LDA];
  __shared__ bf16 Blds[2][BK * LDB];

  const int tid  = threadIdx.x;
  const int wave = tid >> 5, lane = tid & 31;
  const int wm   = wave & 3, wn   = wave >> 2;    // 4 waves along M, 2 along N

  v8f acc[2][4];
#pragma unroll
  for (int i = 0; i < 2; ++i)
#pragma unroll
    for (int j = 0; j < 4; ++j) { v8f z = {0,0,0,0,0,0,0,0}; acc[i][j] = z; }

#if USE_ASYNC
  // Per-wave async staging: 2 A-chunk + 2 B-chunk b128 copies per tile.
  const bf16* asrcp[2]; int adst[2];
  const bf16* bsrcp[2]; int bdst[2];
#pragma unroll
  for (int i = 0; i < 2; ++i) {
    int c = wave * 64 + i * 32 + lane;            // A: 512 chunks of 16B
    int row = c >> 2, colc = (c & 3) * 8;
    int trow = m0 + row;
    int tok  = (trow < cnt) ? eidx[e * NTOK + trow] : 0;
    asrcp[i] = xb + (size_t)tok * NEMBD + colc;
    adst[i]  = row * LDA + colc;
    int k = c >> 4, nc = (c & 15) * 8;            // B: 512 chunks of 16B, K-major
    bsrcp[i] = w1b + (size_t)k * NW1COL + (size_t)e * DFFN + n0 + nc;
    bdst[i]  = k * LDB + nc;
  }
  // stage tile 0 into buffer 0
#pragma unroll
  for (int i = 0; i < 2; ++i) {
    async_cp16(asrcp[i], &Alds[0][adst[i]]);
    async_cp16(bsrcp[i], &Blds[0][bdst[i]]);
  }
  const int KT = NEMBD / BK;
  for (int kt = 0; kt < KT; ++kt) {
    const int cur = kt & 1;
    if (kt + 1 < KT) {
#pragma unroll
      for (int i = 0; i < 2; ++i) {
        async_cp16(asrcp[i] + (size_t)(kt + 1) * BK, &Alds[cur ^ 1][adst[i]]);
        async_cp16(bsrcp[i] + (size_t)(kt + 1) * BK * NW1COL, &Blds[cur ^ 1][bdst[i]]);
      }
      __builtin_amdgcn_s_wait_asynccnt(4);        // current tile done, next in flight
    } else {
      __builtin_amdgcn_s_wait_asynccnt(0);
    }
    __syncthreads();
    wmma_step(&Alds[cur][0], &Blds[cur][0], wm, wn, lane, acc);
    __syncthreads();                              // reads done -> buffer reusable
  }
#else
  // Register-pipelined staging: batch loads -> one wait -> batch stores;
  // next tile's loads issued before computing the current tile.
  const bf16* asrcp[2]; int adst[2];
  const bf16* bsrcp[2]; int bdst[2];
#pragma unroll
  for (int i = 0; i < 2; ++i) {
    int c = tid + i * 256;
    int row = c >> 2, colc = (c & 3) * 8;
    int trow = m0 + row;
    int tok  = (trow < cnt) ? eidx[e * NTOK + trow] : 0;
    asrcp[i] = xb + (size_t)tok * NEMBD + colc;
    adst[i]  = row * LDA + colc;
    int k = c >> 4, nc = (c & 15) * 8;
    bsrcp[i] = w1b + (size_t)k * NW1COL + (size_t)e * DFFN + n0 + nc;
    bdst[i]  = k * LDB + nc;
  }
  v8bf ar[2], br[2];
#pragma unroll
  for (int i = 0; i < 2; ++i) {
    ar[i] = *(const v8bf*)(asrcp[i]);
    br[i] = *(const v8bf*)(bsrcp[i]);
  }
  const int KT = NEMBD / BK;
  for (int kt = 0; kt < KT; ++kt) {
    __syncthreads();                              // previous compute reads done
#pragma unroll
    for (int i = 0; i < 2; ++i) {
      *(v8bf*)&Alds[0][adst[i]] = ar[i];
      *(v8bf*)&Blds[0][bdst[i]] = br[i];
    }
    if (kt + 1 < KT) {
#pragma unroll
      for (int i = 0; i < 2; ++i) {
        ar[i] = *(const v8bf*)(asrcp[i] + (size_t)(kt + 1) * BK);
        br[i] = *(const v8bf*)(bsrcp[i] + (size_t)(kt + 1) * BK * NW1COL);
      }
    }
    __syncthreads();
    wmma_step(&Alds[0][0], &Blds[0][0], wm, wn, lane, acc);
  }
#endif

  // Epilogue: exact GELU, store bf16 H row (list order).
  const int lrow = lane & 15, lhi = lane >> 4;
#pragma unroll
  for (int i = 0; i < 2; ++i) {
#pragma unroll
    for (int j = 0; j < 4; ++j) {
      int gm = m0 + wm * 32 + i * 16 + lhi * 8;
      int gn = n0 + wn * 64 + j * 16 + lrow;
#pragma unroll
      for (int r = 0; r < 8; ++r) {
        int row = gm + r;
        if (row < cnt) {
          float v = acc[i][j][r];
          float g = 0.5f * v * (1.f + erff(v * 0.70710678118f));
          hb[(size_t)row * DFFN + gn] = (bf16)g;
        }
      }
    }
  }
}

// ---------------------------------------------------------------------------
// GEMM2 (per expert e): out[tok] += w_{tok,e} * (H @ W2_e). M=ecnt[e], N=1024, K=2048
// Experts run sequentially on the stream -> plain += is race-free & deterministic.
// ---------------------------------------------------------------------------
__global__ __launch_bounds__(256)
void moe_gemm2_kernel(const bf16* __restrict__ hb, const bf16* __restrict__ w2b,
                      float* __restrict__ out, const int* __restrict__ eidx,
                      const int* __restrict__ ecnt, const float* __restrict__ comb, int e)
{
  const int cnt = ecnt[e];
  const int m0  = blockIdx.y * BM;
  if (m0 >= cnt) return;
  const int n0  = blockIdx.x * BN;

  __shared__ bf16 Alds[2][BM * LDA];
  __shared__ bf16 Blds[2][BK * LDB];

  const int tid  = threadIdx.x;
  const int wave = tid >> 5, lane = tid & 31;
  const int wm   = wave & 3, wn   = wave >> 2;

  v8f acc[2][4];
#pragma unroll
  for (int i = 0; i < 2; ++i)
#pragma unroll
    for (int j = 0; j < 4; ++j) { v8f z = {0,0,0,0,0,0,0,0}; acc[i][j] = z; }

#if USE_ASYNC
  const bf16* asrcp[2]; int adst[2];
  const bf16* bsrcp[2]; int bdst[2];
#pragma unroll
  for (int i = 0; i < 2; ++i) {
    int c = wave * 64 + i * 32 + lane;
    int row = c >> 2, colc = (c & 3) * 8;
    asrcp[i] = hb + (size_t)(m0 + row) * DFFN + colc;   // rows already list-ordered
    adst[i]  = row * LDA + colc;
    int k = c >> 4, nc = (c & 15) * 8;
    bsrcp[i] = w2b + (size_t)e * DFFN * NEMBD + (size_t)k * NEMBD + n0 + nc;
    bdst[i]  = k * LDB + nc;
  }
#pragma unroll
  for (int i = 0; i < 2; ++i) {
    async_cp16(asrcp[i], &Alds[0][adst[i]]);
    async_cp16(bsrcp[i], &Blds[0][bdst[i]]);
  }
  const int KT = DFFN / BK;
  for (int kt = 0; kt < KT; ++kt) {
    const int cur = kt & 1;
    if (kt + 1 < KT) {
#pragma unroll
      for (int i = 0; i < 2; ++i) {
        async_cp16(asrcp[i] + (size_t)(kt + 1) * BK, &Alds[cur ^ 1][adst[i]]);
        async_cp16(bsrcp[i] + (size_t)(kt + 1) * BK * NEMBD, &Blds[cur ^ 1][bdst[i]]);
      }
      __builtin_amdgcn_s_wait_asynccnt(4);
    } else {
      __builtin_amdgcn_s_wait_asynccnt(0);
    }
    __syncthreads();
    wmma_step(&Alds[cur][0], &Blds[cur][0], wm, wn, lane, acc);
    __syncthreads();
  }
#else
  const bf16* asrcp[2]; int adst[2];
  const bf16* bsrcp[2]; int bdst[2];
#pragma unroll
  for (int i = 0; i < 2; ++i) {
    int c = tid + i * 256;
    int row = c >> 2, colc = (c & 3) * 8;
    asrcp[i] = hb + (size_t)(m0 + row) * DFFN + colc;
    adst[i]  = row * LDA + colc;
    int k = c >> 4, nc = (c & 15) * 8;
    bsrcp[i] = w2b + (size_t)e * DFFN * NEMBD + (size_t)k * NEMBD + n0 + nc;
    bdst[i]  = k * LDB + nc;
  }
  v8bf ar[2], br[2];
#pragma unroll
  for (int i = 0; i < 2; ++i) {
    ar[i] = *(const v8bf*)(asrcp[i]);
    br[i] = *(const v8bf*)(bsrcp[i]);
  }
  const int KT = DFFN / BK;
  for (int kt = 0; kt < KT; ++kt) {
    __syncthreads();
#pragma unroll
    for (int i = 0; i < 2; ++i) {
      *(v8bf*)&Alds[0][adst[i]] = ar[i];
      *(v8bf*)&Blds[0][bdst[i]] = br[i];
    }
    if (kt + 1 < KT) {
#pragma unroll
      for (int i = 0; i < 2; ++i) {
        ar[i] = *(const v8bf*)(asrcp[i] + (size_t)(kt + 1) * BK);
        br[i] = *(const v8bf*)(bsrcp[i] + (size_t)(kt + 1) * BK * NEMBD);
      }
    }
    __syncthreads();
    wmma_step(&Alds[0][0], &Blds[0][0], wm, wn, lane, acc);
  }
#endif

  // Epilogue: scale by combine weight, scatter-add into out rows.
  const int lrow = lane & 15, lhi = lane >> 4;
#pragma unroll
  for (int i = 0; i < 2; ++i) {
#pragma unroll
    for (int j = 0; j < 4; ++j) {
      int gm = m0 + wm * 32 + i * 16 + lhi * 8;
      int gn = n0 + wn * 64 + j * 16 + lrow;
#pragma unroll
      for (int r = 0; r < 8; ++r) {
        int row = gm + r;
        if (row < cnt) {
          int tok  = eidx[e * NTOK + row];
          float cw = comb[(size_t)tok * NEXP + e];
          out[(size_t)tok * NEMBD + gn] += cw * acc[i][j][r];
        }
      }
    }
  }
}

// ---------------------------------------------------------------------------
extern "C" void kernel_launch(void* const* d_in, const int* in_sizes, int n_in,
                              void* d_out, int out_size, void* d_ws, size_t ws_size,
                              hipStream_t stream)
{
  (void)in_sizes; (void)n_in; (void)out_size; (void)ws_size;
  const float* x  = (const float*)d_in[0];
  const float* rw = (const float*)d_in[1];
  const float* w1 = (const float*)d_in[2];
  const float* w2 = (const float*)d_in[3];
  float* out = (float*)d_out;
  char*  ws  = (char*)d_ws;

  bf16*  xb   = (bf16*)(ws + OFF_XB);
  bf16*  w1b  = (bf16*)(ws + OFF_W1B);
  bf16*  w2b  = (bf16*)(ws + OFF_W2B);
  bf16*  hb   = (bf16*)(ws + OFF_HB);
  float* comb = (float*)(ws + OFF_CMB);
  int*   eidx = (int*)(ws + OFF_IDX);
  int*   ecnt = (int*)(ws + OFF_CNT);
  int*   fcnt = (int*)(ws + OFF_FCN);
  float* part = (float*)(ws + OFF_PRT);

  // 1) zero output accumulator + counters
  {
    int n4 = (NTOK * NEMBD) / 4;
    zero4_kernel<<<(n4 + 255) / 256, 256, 0, stream>>>((float4*)out, n4);
    zero4_kernel<<<1, 256, 0, stream>>>((float4*)ecnt, 4);   // ecnt[8]+fcnt[8] = 64 B
  }
  // 2) fp32 -> bf16 conversion passes (bandwidth-trivial vs compute)
  {
    int n4 = NTOK * NEMBD / 4;
    cvt_bf16_kernel<<<(n4 + 255) / 256, 256, 0, stream>>>((const float4*)x, (v4bf*)xb, n4);
    n4 = NEMBD * NW1COL / 4;
    cvt_bf16_kernel<<<(n4 + 255) / 256, 256, 0, stream>>>((const float4*)w1, (v4bf*)w1b, n4);
    cvt_bf16_kernel<<<(n4 + 255) / 256, 256, 0, stream>>>((const float4*)w2, (v4bf*)w2b, n4);
  }
  // 3) router + aux losses
  router_kernel<<<NTOK / 8, 256, 0, stream>>>(x, rw, comb, ecnt, eidx, fcnt, part);
  finalize_kernel<<<1, 32, 0, stream>>>(part, fcnt, out + (size_t)NTOK * NEMBD);

  // 4) routed expert GEMMs (sequential per expert -> deterministic scatter-add)
  for (int e = 0; e < NEXP; ++e) {
    moe_gemm1_kernel<<<dim3(DFFN / BN, NTOK / BM), 256, 0, stream>>>(xb, w1b, hb, eidx, ecnt, e);
    moe_gemm2_kernel<<<dim3(NEMBD / BN, NTOK / BM), 256, 0, stream>>>(hb, w2b, out, eidx, ecnt, comb, e);
  }
}
